// MultiFeatureTransformer_9809705305084
// MI455X (gfx1250) — matrix operand order repaired and checked by reading, not compiled, not run on hardware
//
#include <hip/hip_runtime.h>

typedef __attribute__((ext_vector_type(16))) _Float16 v16h;
typedef __attribute__((ext_vector_type(8)))  _Float16 v8h;
typedef __attribute__((ext_vector_type(2)))  _Float16 h2v;
typedef __attribute__((ext_vector_type(8)))  float    v8f;
typedef __attribute__((ext_vector_type(4)))  float    f32x4;
typedef int v4i32 __attribute__((vector_size(16)));   // matches builtin param spelling

#define BTILE 256
#define NEG_SLOPE 0.05f

// ---- async global->LDS path (ASYNCcnt), guarded for toolchain differences ----
#if defined(__has_builtin)
#  if __has_builtin(__builtin_amdgcn_global_load_async_to_lds_b128) && \
      __has_builtin(__builtin_amdgcn_global_load_async_to_lds_b32)
#    define USE_ASYNC 1
#  endif
#endif
#ifndef USE_ASYNC
#  define USE_ASYNC 0
#endif

__device__ __forceinline__ void async_b128(const void* g, void* l) {
#if USE_ASYNC
    __builtin_amdgcn_global_load_async_to_lds_b128(
        (__attribute__((address_space(1))) v4i32*)(void*)g,
        (__attribute__((address_space(3))) v4i32*)l, 0, 0);
#else
    *(f32x4*)l = *(const f32x4*)g;
#endif
}

__device__ __forceinline__ void async_b32(const void* g, void* l) {
#if USE_ASYNC
    __builtin_amdgcn_global_load_async_to_lds_b32(
        (__attribute__((address_space(1))) int*)(void*)g,
        (__attribute__((address_space(3))) int*)l, 0, 0);
#else
    *(float*)l = *(const float*)g;
#endif
}

template <int N>
__device__ __forceinline__ void wait_async() {
#if USE_ASYNC
#  if __has_builtin(__builtin_amdgcn_s_wait_asynccnt)
    __builtin_amdgcn_s_wait_asynccnt((short)N);
#  else
    asm volatile("s_wait_asynccnt %0" :: "i"(N) : "memory");
#  endif
#endif
}

// One block: feature f, 256 batch rows. 8 waves x 2 slabs x (16 rows x 64 g) via WMMA.
__global__ __launch_bounds__(256) void mft_kernel(
    const float* __restrict__ x,   // (B,1,64)
    const float* __restrict__ W1,  // (64,64)
    const float* __restrict__ b1,  // (64,64)
    const float* __restrict__ W2,  // (64,64,64) [f][g][h]
    const float* __restrict__ b2,  // (64,64)
    const float* __restrict__ W3,  // (64,64)
    const float* __restrict__ b3,  // (64,)
    float* __restrict__ y)         // (B,1,64)
{
    __shared__ h2v   sW1h[32], sB1h[32];            // f16 pairs for packed phase 2
    __shared__ float sB2[64], sW3[64];
    __shared__ float sX[BTILE];
    __shared__ __align__(16) float    sW2raw[4096]; // raw W2[f] (f32), async-filled
    __shared__ __align__(32) _Float16 sBfrag[4096]; // [kstep(2)][nt(4)][lane(32)][j(16)]
    __shared__ __align__(16) _Float16 sH1[BTILE * 64]; // [row][h], row stride 128B

    const int t  = threadIdx.x;
    const int f  = blockIdx.x & 63;
    const int b0 = (int)(blockIdx.x >> 6) * BTILE;
    const float* __restrict__ W2f = W2 + (size_t)f * 4096;

    // ---- phase 0: issue async copies. x first, then W2 (in-order completion) ----
    async_b32(&x[(size_t)(b0 + t) * 64 + f], &sX[t]);
#pragma unroll
    for (int i = 0; i < 4; ++i) {
        int off = t * 4 + i * 1024;  // floats; 256 thr x 16B x 4 = 16KB
        async_b128(W2f + off, &sW2raw[off]);
    }

    // small weights via regular loads (latency hidden behind asyncs)
    if      (t <  64) ((_Float16*)sW1h)[t] = (_Float16)W1[f * 64 + t];
    else if (t < 128) ((_Float16*)sB1h)[t - 64] = (_Float16)b1[f * 64 + (t - 64)];
    else if (t < 192) sB2[t - 128] = b2[f * 64 + (t - 128)];
    else              sW3[t - 192] = W3[f * 64 + (t - 192)];
    const float b3f = b3[f];

    wait_async<4>();          // release x tile only; W2 (4 asyncs) still in flight
    __syncthreads();

    // ---- phase 2: h1 = leaky(x*W1 + b1) in packed f16 while W2 streams in ----
    // leaky(v) == max(v, slope*v) since slope > 0
    {
        const h2v slope2 = {(_Float16)NEG_SLOPE, (_Float16)NEG_SLOPE};
        h2v* __restrict__ sH1p = (h2v*)sH1;
        for (int e2 = t; e2 < BTILE * 32; e2 += 256) {
            int r = e2 >> 5, hp = e2 & 31;      // r uniform per wave; hp == lane
            _Float16 xh = (_Float16)sX[r];
            h2v xx = {xh, xh};
            h2v v = xx * sW1h[hp] + sB1h[hp];   // v_pk_fma_f16
            v = __builtin_elementwise_max(v, v * slope2);
            sH1p[r * 32 + hp] = v;
        }
    }

    wait_async<0>();          // W2 raw tile complete
    __syncthreads();

    // ---- phase 2b: permute W2 -> f16 B-fragment lane order ----
    // fragment value j of lane L for (kstep,nt) is W2[f][g][h],
    //   g = nt*16 + (L&15),  h = kstep*32 + ((L>>4)&1)*8 + (j&7) + (j>>3)*16
    for (int i = t; i < 4096; i += 256) {
        int kstep = i >> 11, nt = (i >> 9) & 3, ln = (i >> 4) & 31, j = i & 15;
        int g = nt * 16 + (ln & 15);
        int h = kstep * 32 + ((ln >> 4) & 1) * 8 + (j & 7) + (j >> 3) * 16;
        sBfrag[i] = (_Float16)sW2raw[g * 64 + h];
    }
    __syncthreads();

    // ---- phase 3/4: WMMA + epilogue, two 128-row slabs per block ----
    const int lane = t & 31;
    const int w    = t >> 5;
    const int m    = lane & 15;
    const int hv   = (lane >> 4) & 1;
    const int kb   = hv * 8;  // A-frag per-lane K base within 32-K block

#pragma unroll
    for (int slab = 0; slab < 2; ++slab) {
        const _Float16* __restrict__ h1row = &sH1[(slab * 128 + w * 16 + m) * 64];
        v8f acc[4] = {};

#pragma unroll
        for (int kstep = 0; kstep < 2; ++kstep) {
            // 16-bit A layout: VGPR0-3 = K (kb..kb+7), VGPR4-7 = K (kb+16..kb+23)
            v8h lo = *(const v8h*)(h1row + kstep * 32 + kb);
            v8h hi = *(const v8h*)(h1row + kstep * 32 + kb + 16);
            v16h a;
#pragma unroll
            for (int j = 0; j < 8; ++j) { a[j] = lo[j]; a[j + 8] = hi[j]; }

#pragma unroll
            for (int nt = 0; nt < 4; ++nt) {
                v16h bf = *(const v16h*)&sBfrag[((kstep * 4 + nt) * 32 + lane) * 16];
                acc[nt] = __builtin_amdgcn_wmma_f32_16x16x32_f16(
                    false, a, false, bf, (short)0, acc[nt], false, false);
            }
        }

        // epilogue: +b2, leaky(max), dot W3, 16-lane butterfly reduce, +b3
        // C/D layout: VGPR r holds row (w*16 + r + hv*8), column g = nt*16 + m
#pragma unroll
        for (int r = 0; r < 8; ++r) {
            float p = 0.f;
#pragma unroll
            for (int nt = 0; nt < 4; ++nt) {
                int g = nt * 16 + m;
                float v = acc[nt][r] + sB2[g];
                v = fmaxf(v, v * NEG_SLOPE);
                p = fmaf(v, sW3[g], p);
            }
            p += __shfl_xor(p, 1, 32);
            p += __shfl_xor(p, 2, 32);
            p += __shfl_xor(p, 4, 32);
            p += __shfl_xor(p, 8, 32);
            if (m == 0) {
                int rr = slab * 128 + w * 16 + r + hv * 8;
                y[(size_t)(b0 + rr) * 64 + f] = p + b3f;
            }
        }
    }
}

extern "C" void kernel_launch(void* const* d_in, const int* in_sizes, int n_in,
                              void* d_out, int out_size, void* d_ws, size_t ws_size,
                              hipStream_t stream) {
    const float* x  = (const float*)d_in[0];
    const float* W1 = (const float*)d_in[1];
    const float* b1 = (const float*)d_in[2];
    const float* W2 = (const float*)d_in[3];
    const float* b2 = (const float*)d_in[4];
    const float* W3 = (const float*)d_in[5];
    const float* b3 = (const float*)d_in[6];
    float* y = (float*)d_out;

    const int B = in_sizes[0] / 64;            // x is (B,1,64)
    const int nblocks = (B / BTILE) * 64;      // 64 features * batch tiles
    mft_kernel<<<nblocks, 256, 0, stream>>>(x, W1, b1, W2, b2, W3, b3, y);
}
